// LSS_dev_91018946937221
// MI455X (gfx1250) — compile-verified, matrix operand order checked by reading, not run on hardware
//
#include <hip/hip_runtime.h>
#include <hip/hip_bf16.h>

// ---------------- problem constants ----------------
#define NB       4            // batch
#define NCAM     6            // cameras
#define CIN      512
#define FH_      16
#define FW_      44
#define DBINS    59
#define CTRANS   64
#define COUT     128          // 123 padded to 128
#define PIXIMG   (FH_*FW_)    // 704
#define NPIX     (NB*NCAM*PIXIMG)      // 16896
#define MTILES   (NPIX/16)             // 1056
#define KSTEPS   (CIN/32)              // 16
#define NTILES   (COUT/16)             // 8
#define OUTELEM  (NB*CTRANS*128*128)   // 4194304
// packed weight: KSTEPS*NTILES tiles, each 32x16 bf16 = 256 dwords, lane-major
#define WPK_DW   (KSTEPS*NTILES*256)   // 32768 dwords = 128KB

typedef __attribute__((ext_vector_type(16))) __bf16 v16bf;
typedef __attribute__((ext_vector_type(8)))  float  v8f;
typedef __attribute__((ext_vector_type(4))) unsigned u32x4;
typedef __attribute__((ext_vector_type(8))) int      i32x8;
typedef __attribute__((ext_vector_type(4))) int      i32x4;

__device__ __forceinline__ unsigned bf1(float f) {
    unsigned u = __builtin_bit_cast(unsigned, f);
    return (u + 0x7FFFu + ((u >> 16) & 1u)) >> 16;   // round-to-nearest-even
}
__device__ __forceinline__ unsigned bfpack(float lo, float hi) {
#if defined(__has_builtin) && __has_builtin(__builtin_amdgcn_cvt_pk_bf16_f32)
    typedef __attribute__((ext_vector_type(2))) __bf16 v2bf;
    v2bf r = __builtin_amdgcn_cvt_pk_bf16_f32(lo, hi);
    return __builtin_bit_cast(unsigned, r);
#else
    return bf1(lo) | (bf1(hi) << 16);
#endif
}

// ---------------- kernel: zero output ----------------
__global__ void k_zero(float* __restrict__ out, int n) {
    for (int i = blockIdx.x * blockDim.x + threadIdx.x; i < n; i += gridDim.x * blockDim.x)
        out[i] = 0.0f;
}

// ---------------- kernel: pack W_depth (123x512 f32) -> bf16 B-layout ----------------
// dword q: j = q&7 (VGPR), L = (q>>3)&31 (lane), nt = (q>>8)&7, ks = q>>11
// lane L holds n = nt*16 + (L&15); K-half = (L>>4)*8
// VGPR j<4 -> k = 2j, 2j+1 ; j>=4 -> k = 16+2(j-4), +1   (all + khalf + ks*32)
__global__ void k_packw(const float* __restrict__ Wd, unsigned* __restrict__ wpk) {
    int q = blockIdx.x * blockDim.x + threadIdx.x;
    if (q >= WPK_DW) return;
    int j  = q & 7;
    int L  = (q >> 3) & 31;
    int nt = (q >> 8) & 7;
    int ks = q >> 11;
    int n  = nt * 16 + (L & 15);
    int kk = ks * 32 + ((L >> 4) * 8) + (j < 4 ? 2 * j : 16 + 2 * (j - 4));
    float w0 = (n < 123) ? Wd[n * CIN + kk]     : 0.0f;
    float w1 = (n < 123) ? Wd[n * CIN + kk + 1] : 0.0f;
    wpk[q] = bfpack(w0, w1);
}

// ---------------- kernel: per-camera combined transforms ----------------
__device__ void inv3(const float* m, float* o) {
    float a=m[0],b=m[1],c=m[2],d=m[3],e=m[4],f=m[5],g=m[6],h=m[7],i=m[8];
    float A = e*i - f*h, B = -(d*i - f*g), C = d*h - e*g;
    float r = 1.0f / (a*A + b*B + c*C);
    o[0]=A*r;            o[1]=-(b*i-c*h)*r;   o[2]=(b*f-c*e)*r;
    o[3]=B*r;            o[4]=(a*i-c*g)*r;    o[5]=-(a*f-c*d)*r;
    o[6]=C*r;            o[7]=-(a*h-b*g)*r;   o[8]=(a*e-b*d)*r;
}
__global__ void k_cam(const float* __restrict__ rots, const float* __restrict__ trans,
                      const float* __restrict__ intr, const float* __restrict__ prots,
                      const float* __restrict__ ptrans, float* __restrict__ cam) {
    int bn = blockIdx.x * blockDim.x + threadIdx.x;
    if (bn >= NB * NCAM) return;
    float Minv[9], Ki[9], Cm[9];
    inv3(prots + bn * 9, Minv);
    inv3(intr  + bn * 9, Ki);
    const float* R = rots + bn * 9;
    for (int r = 0; r < 3; ++r)
        for (int c = 0; c < 3; ++c)
            Cm[r*3+c] = R[r*3+0]*Ki[0*3+c] + R[r*3+1]*Ki[1*3+c] + R[r*3+2]*Ki[2*3+c];
    float* o = cam + bn * 24;
    for (int i = 0; i < 9; ++i) { o[i] = Minv[i]; o[9 + i] = Cm[i]; }
    for (int i = 0; i < 3; ++i) { o[18 + i] = ptrans[bn*3 + i]; o[21 + i] = trans[bn*3 + i]; }
}

// ---------------- kernel: WMMA GEMM  feat[p,128] = x[p,:512] @ W^T ----------------
// 256 threads = 8 waves; wave w handles M-tile (blockIdx*8 + w): 16 pixels x 128 outs.
// Whole packed W (128KB bf16) staged once per workgroup into LDS via the Tensor
// Data Mover (TENSOR_LOAD_TO_LDS, TENSORcnt), WGP LDS budget 320KB.
__global__ __launch_bounds__(256) void k_gemm(const float* __restrict__ x,
                                              const uint4* __restrict__ wpk,
                                              float* __restrict__ feat) {
    extern __shared__ uint4 lw[];                      // 8192 uint4 = 128KB
    int tid = threadIdx.x;

#if defined(__has_builtin) && __has_builtin(__builtin_amdgcn_tensor_load_to_lds)
    // --- TDM: one DMA of the whole packed-W tile (32768 dwords, contiguous) ---
    if (tid == 0) {
        unsigned      ldsoff = (unsigned)(__SIZE_TYPE__)(void*)lw;       // LDS byte offset
        unsigned long gaddr  = (unsigned long)(const void*)wpk;          // global byte addr
        // D# group0: count=1 | lds_addr | global_addr[56:0] | type=2
        u32x4 g0;
        g0.x = 1u;                                       // count=1, user mode
        g0.y = ldsoff;                                   // lds_addr (bytes)
        g0.z = (unsigned)(gaddr & 0xFFFFFFFFu);          // global_addr[31:0]
        g0.w = (unsigned)((gaddr >> 32) & 0x1FFFFFFu)    // global_addr[56:32]
               | 0x80000000u;                            // type=2 ("image")
        // D# group1: data_size=4B(code 2); tensor_dim0=32768; tensor_dim1=1;
        //            tile_dim0=32768 (1-row tile); dim0 stride=32768
        i32x8 g1;
        g1[0] = 0x00020000;                              // data_size=2<<16
        g1[1] = (int)0x80000000u;                        // tensor_dim0[15:0]=0x8000 <<16
        g1[2] = 0x00010000;                              // dim0[31:16]=0 | tensor_dim1=1 <<16
        g1[3] = (int)0x80000000u;                        // dim1[31:16]=0 | tile_dim0=0x8000 <<16
        g1[4] = 0;                                       // tile_dim1=0 (unused), tile_dim2=0
        g1[5] = 32768;                                   // tensor_dim0_stride[31:0]
        g1[6] = 0;
        g1[7] = 0;
        i32x4 gz4 = (i32x4){0, 0, 0, 0};                 // groups 2/3 unused (<=2D)
        i32x8 gz8 = (i32x8){0, 0, 0, 0, 0, 0, 0, 0};     // extra group (clang-23 form)
        __builtin_amdgcn_tensor_load_to_lds(g0, g1, gz4, gz4, gz8, 0);
    }
    __builtin_amdgcn_s_wait_tensorcnt(0);                // issuing wave: DMA complete
    __syncthreads();                                     // publish LDS to all waves
#else
    for (int i = tid; i < WPK_DW / 4; i += 256) lw[i] = wpk[i];
    __syncthreads();
#endif

    int wave = tid >> 5, lane = tid & 31;
    int mtile  = blockIdx.x * 8 + wave;                // 0..1055 (grid = 132)
    int bn     = mtile / 44;                           // 44 M-tiles per image (704/16)
    int pixoff = (mtile % 44) * 16;
    int m      = lane & 15;
    int khalf  = (lane >> 4) * 8;
    const float* xb = x + (size_t)bn * (CIN * PIXIMG) + pixoff + m;

    v8f acc[NTILES];
    #pragma unroll
    for (int nt = 0; nt < NTILES; ++nt) acc[nt] = (v8f){0,0,0,0,0,0,0,0};

    for (int ks = 0; ks < KSTEPS; ++ks) {
        int kb = ks * 32 + khalf;
        if (ks + 1 < KSTEPS) {                         // hint next K-step A rows into cache
            __builtin_prefetch(xb + (size_t)(kb + 32) * PIXIMG, 0, 1);
            __builtin_prefetch(xb + (size_t)(kb + 48) * PIXIMG, 0, 1);
        }
        // A fragment: ISA 16-bit A layout, f32 global loads -> bf16 pairs
        union { unsigned u[8]; v16bf v; } a;
        #pragma unroll
        for (int j = 0; j < 8; ++j) {
            int kk = kb + (j < 4 ? 2 * j : 16 + 2 * (j - 4));
            float f0 = xb[(size_t)kk * PIXIMG];
            float f1 = xb[(size_t)(kk + 1) * PIXIMG];
            a.u[j] = bfpack(f0, f1);
        }
        #pragma unroll
        for (int nt = 0; nt < NTILES; ++nt) {
            union { uint4 q[2]; v16bf v; } b;
            int base = (ks * NTILES + nt) * 64 + lane * 2;   // uint4 units
            b.q[0] = lw[base];
            b.q[1] = lw[base + 1];
            acc[nt] = __builtin_amdgcn_wmma_f32_16x16x32_bf16(
                          false, a.v, false, b.v, (short)0, acc[nt], false, false);
        }
    }
    // C layout: VGPR r -> M = r + 8*(lane>=16), N = lane&15 (per n-tile)
    int row0 = mtile * 16;
    #pragma unroll
    for (int nt = 0; nt < NTILES; ++nt) {
        #pragma unroll
        for (int r = 0; r < 8; ++r) {
            int mm = r + ((lane >> 4) << 3);
            int nn = nt * 16 + (lane & 15);
            feat[(size_t)(row0 + mm) * COUT + nn] = acc[nt][r];
        }
    }
}

// ---------------- kernel: softmax + geometry + voxel scatter ----------------
// one 64-thread block per pixel; thread t = output channel
__global__ __launch_bounds__(64) void k_scatter(const float* __restrict__ feat,
                                                const float* __restrict__ bias,
                                                const float* __restrict__ cam,
                                                float* __restrict__ out) {
    __shared__ float red[64];
    __shared__ float dep[64];
    __shared__ float img[64];
    __shared__ int   vbase[DBINS];

    int p  = blockIdx.x;
    int t  = threadIdx.x;
    int bn = p / PIXIMG;
    int pix = p - bn * PIXIMG;
    int h = pix / FW_, w = pix - h * FW_;
    int b = bn / NCAM;

    const float* fp = feat + (size_t)p * COUT;
    float logit = (t < DBINS) ? fp[t] + bias[t] : -1e30f;
    img[t] = fp[DBINS + t] + bias[DBINS + t];          // channels 59..122

    red[t] = logit;  __syncthreads();
    #pragma unroll
    for (int s = 32; s; s >>= 1) { if (t < s) red[t] = fmaxf(red[t], red[t + s]); __syncthreads(); }
    float mx = red[0];  __syncthreads();
    float e = (t < DBINS) ? __expf(logit - mx) : 0.0f;
    red[t] = e;  __syncthreads();
    #pragma unroll
    for (int s = 32; s; s >>= 1) { if (t < s) red[t] += red[t + s]; __syncthreads(); }
    dep[t] = e / red[0];

    if (t < DBINS) {
        const float* C = cam + bn * 24;
        float xs = (float)w * (703.0f / 43.0f);
        float ys = (float)h * 17.0f;
        float d  = (float)t + 1.0f;
        float p0 = xs - C[18], p1 = ys - C[19], p2 = d - C[20];
        float qx = C[0]*p0 + C[1]*p1 + C[2]*p2;
        float qy = C[3]*p0 + C[4]*p1 + C[5]*p2;
        float qz = C[6]*p0 + C[7]*p1 + C[8]*p2;
        float rx = qx * qz, ry = qy * qz, rz = qz;
        float sx = C[ 9]*rx + C[10]*ry + C[11]*rz + C[21];
        float sy = C[12]*rx + C[13]*ry + C[14]*rz + C[22];
        float sz = C[15]*rx + C[16]*ry + C[17]*rz + C[23];
        int gx = (int)floorf((sx + 51.2f) / 0.8f);
        int gy = (int)floorf((sy + 51.2f) / 0.8f);
        int gz = (int)floorf((sz + 10.0f) / 20.0f);
        vbase[t] = (gx >= 0 && gx < 128 && gy >= 0 && gy < 128 && gz == 0)
                       ? (b * (CTRANS * 128 * 128) + gy * 128 + gx) : -1;
    }
    __syncthreads();

    float wimg = img[t];
    float* ob = out + t * (128 * 128);                 // channel offset
    for (int d0 = 0; d0 < DBINS; ++d0) {
        int vb = vbase[d0];
        if (vb >= 0)
            __hip_atomic_fetch_add(&ob[vb], dep[d0] * wimg,
                                   __ATOMIC_RELAXED, __HIP_MEMORY_SCOPE_AGENT);
    }
}

// ---------------- host launcher ----------------
extern "C" void kernel_launch(void* const* d_in, const int* in_sizes, int n_in,
                              void* d_out, int out_size, void* d_ws, size_t ws_size,
                              hipStream_t stream) {
    const float* x      = (const float*)d_in[0];
    const float* rots   = (const float*)d_in[1];
    const float* trans  = (const float*)d_in[2];
    const float* intr   = (const float*)d_in[3];
    const float* prots  = (const float*)d_in[4];
    const float* ptrans = (const float*)d_in[5];
    const float* Wd     = (const float*)d_in[6];
    const float* bd     = (const float*)d_in[7];
    float* out = (float*)d_out;

    char* ws = (char*)d_ws;
    unsigned* wpk = (unsigned*)ws;                       // 131072 B packed bf16 W
    float* cam    = (float*)(ws + 131072);               // 24 cams * 24 f = 2304 B
    float* feat   = (float*)(ws + 133376);               // 16896*128 f32 = 8.65 MB

    k_zero<<<2048, 256, 0, stream>>>(out, OUTELEM);
    k_packw<<<(WPK_DW + 255) / 256, 256, 0, stream>>>(Wd, wpk);
    k_cam<<<1, 32, 0, stream>>>(rots, trans, intr, prots, ptrans, cam);
    k_gemm<<<MTILES / 8, 256, WPK_DW * 4, stream>>>(x, (const uint4*)wpk, feat);
    k_scatter<<<NPIX, 64, 0, stream>>>(feat, bd, cam, out);
}
